// PruneRNN_87359634801232
// MI455X (gfx1250) — matrix operand order repaired and checked by reading, not compiled, not run on hardware
//
#include <hip/hip_runtime.h>
#include <cstdint>
#include <cstddef>

#define S_LEN 2048
#define BATCH 64
#define IN_DIM 256
#define HID 512
#define NWG 8
#define NTHREADS 256

typedef __attribute__((ext_vector_type(16))) _Float16 v16h;
typedef __attribute__((ext_vector_type(8)))  _Float16 v8h;
typedef __attribute__((ext_vector_type(8)))  float    v8f;
typedef __attribute__((ext_vector_type(4)))  float    v4f;
typedef __attribute__((ext_vector_type(4)))  unsigned int v4u;
typedef __attribute__((ext_vector_type(8)))  int      v8i;
typedef __attribute__((ext_vector_type(4)))  int      v4i;

// ---------------------------------------------------------------------------
// WMMA helper: D = A(16x32 f16) * B(32x16 f16) + C(16x16 f32)
// ---------------------------------------------------------------------------
__device__ __forceinline__ v8f wmma16(v16h a, v16h b, v8f c) {
  return __builtin_amdgcn_wmma_f32_16x16x32_f16(
      false, a, false, b, (short)0, c, false, false);
}

// A-fragment (16x32 f16): lane L holds row M=L%16; halfs 0..7 = K [kb,kb+8),
// halfs 8..15 = K [kb+16,kb+24), kb = k0 + 8*(L>=16).
__device__ __forceinline__ v16h load_a_f16(const _Float16* __restrict__ row, int kb) {
  v8h lo = *(const v8h*)(row + kb);
  v8h hi = *(const v8h*)(row + kb + 16);
  v16h a;
#pragma unroll
  for (int i = 0; i < 8; ++i) { a[i] = lo[i]; a[i + 8] = hi[i]; }
  return a;
}

// B-fragment from pre-swizzled buffer: frag[((tj*ktiles+tk)*32+lane)*16+i]
__device__ __forceinline__ v16h load_b_frag(const _Float16* __restrict__ frag,
                                            int tj, int kt, int ktiles, int lane) {
  return *(const v16h*)(frag + (((size_t)(tj * ktiles + kt) * 32 + lane) * 16));
}

// ---------------------------------------------------------------------------
// TDM: 1-D Global -> LDS DMA (bytes must be a multiple of 8).
// D# built per CDNA5 ISA: group0 = {count/lds/global/type},
// group1 = {data_size=8B, tensor_dim0=tile_dim0=n8, dims1=1, stride0=n8}.
// Tracked by TENSORcnt; wave-uniform arguments -> SGPR descriptor groups.
// This toolchain's builtin is the 6-arg variant:
//   (uint32x4 g0, int32x8 g1, int32x4 g2, int32x4 g3, int32x8 gx, i32 cpol)
// ---------------------------------------------------------------------------
__device__ __forceinline__ void tdm_load_1d(unsigned lds_byte_off,
                                            const void* gsrc, unsigned bytes) {
  unsigned long long ga = (unsigned long long)(uintptr_t)gsrc;
  unsigned n8 = bytes >> 3;  // elements of 8 bytes
  v4u g0;
  g0[0] = 1u;                                       // count=1 (valid), user mode
  g0[1] = lds_byte_off;                             // lds_addr
  g0[2] = (unsigned)(ga & 0xFFFFFFFFu);             // global_addr[31:0]
  g0[3] = (unsigned)((ga >> 32) & 0x01FFFFFFu)      // global_addr[56:32]
          | (2u << 30);                             // type=2 ("image")
  v8i g1;
  g1[0] = (3 << 16);                                // wg_mask=0, data_size=3 (8B)
  g1[1] = (int)((n8 & 0xFFFFu) << 16);              // tensor_dim0[15:0]
  g1[2] = (int)((n8 >> 16) & 0xFFFFu) | (1 << 16);  // tensor_dim0[31:16] | tensor_dim1=1
  g1[3] = (int)((n8 & 0xFFFFu) << 16);              // tile_dim0 = n8
  g1[4] = 1;                                        // tile_dim1=1, tile_dim2=0
  g1[5] = (int)n8;                                  // tensor_dim0_stride[31:0]
  g1[6] = 0;
  g1[7] = 0;
  v4i gz4 = {0, 0, 0, 0};
  v8i gz8 = {0, 0, 0, 0, 0, 0, 0, 0};
  __builtin_amdgcn_tensor_load_to_lds(g0, g1, gz4, gz4, gz8, 0);
}

// ---------------------------------------------------------------------------
// Prep: mask weights, cast to f16, scatter into WMMA B-fragment layout.
// ---------------------------------------------------------------------------
__global__ void prune_prep_weights(const float* __restrict__ W,
                                   const int* __restrict__ mask,
                                   _Float16* __restrict__ frag, int K) {
  int idx = blockIdx.x * blockDim.x + threadIdx.x;
  int total = HID * K;
  if (idx >= total) return;
  int j = idx / K;
  int k = idx - j * K;
  float v = W[idx] * (float)mask[idx];
  int tj = j >> 4, tk = k >> 5;
  int jj = j & 15, kk = k & 31;
  int lane = jj + ((kk & 8) ? 16 : 0);
  int i = (kk & 7) + ((kk & 16) ? 8 : 0);
  frag[(((size_t)(tj * (K >> 5) + tk) * 32 + lane) * 16) + i] = (_Float16)v;
}

// Bias sums, zero hidden-state double buffers, reset grid barrier.
__global__ void prune_prep_misc(const float* __restrict__ bih0,
                                const float* __restrict__ bhh0,
                                const float* __restrict__ bih1,
                                const float* __restrict__ bhh1,
                                float* __restrict__ bias0, float* __restrict__ bias1,
                                _Float16* __restrict__ h0, _Float16* __restrict__ h1,
                                unsigned* bar, unsigned* gen) {
  int idx = blockIdx.x * blockDim.x + threadIdx.x;
  if (idx < HID) {
    bias0[idx] = bih0[idx] + bhh0[idx];
    bias1[idx] = bih1[idx] + bhh1[idx];
  }
  const int n = 2 * BATCH * HID;
  if (idx < n) { h0[idx] = (_Float16)0.0f; h1[idx] = (_Float16)0.0f; }
  if (idx == 0) { *bar = 0u; *gen = 0u; }
}

// ---------------------------------------------------------------------------
// Parallel GEMM: xb0[s,b,j] = sum_k x[s,b,k]*Wih0m[j,k] + bih0[j]+bhh0[j]
// ---------------------------------------------------------------------------
__global__ __launch_bounds__(NTHREADS) void prune_xb0_gemm(
    const float* __restrict__ x, const _Float16* __restrict__ fWih0,
    const float* __restrict__ bias0, _Float16* __restrict__ xb0) {
  const int s = blockIdx.x;
  const int lane = threadIdx.x & 31;
  const int w = threadIdx.x >> 5;
  const int m = w & 3;
  const int KT = IN_DIM / 32;  // 8

  const int M = m * 16 + (lane & 15);
  const float* xrow = x + ((size_t)s * BATCH + M) * IN_DIM;

  v16h afr[KT];
#pragma unroll
  for (int kt = 0; kt < KT; ++kt) {
    int kb = kt * 32 + ((lane & 16) ? 8 : 0);
    v4f a0 = *(const v4f*)(xrow + kb);
    v4f a1 = *(const v4f*)(xrow + kb + 4);
    v4f a2 = *(const v4f*)(xrow + kb + 16);
    v4f a3 = *(const v4f*)(xrow + kb + 20);
    v16h a;
#pragma unroll
    for (int i = 0; i < 4; ++i) {
      a[i]      = (_Float16)a0[i];
      a[4 + i]  = (_Float16)a1[i];
      a[8 + i]  = (_Float16)a2[i];
      a[12 + i] = (_Float16)a3[i];
    }
    afr[kt] = a;
  }

  const int rb = (lane & 16) ? 8 : 0;
  for (int nn = 0; nn < 16; ++nn) {
    int tj = (w >> 2) * 16 + nn;
    v8f acc = {};
#pragma unroll
    for (int kt = 0; kt < KT; ++kt) {
      v16h b = load_b_frag(fWih0, tj, kt, KT, lane);
      acc = wmma16(afr[kt], b, acc);
    }
    int j = tj * 16 + (lane & 15);
    float bj = bias0[j];
#pragma unroll
    for (int r = 0; r < 8; ++r) {
      int bi = m * 16 + rb + r;
      xb0[((size_t)s * BATCH + bi) * HID + j] = (_Float16)(acc[r] + bj);
    }
  }
}

// ---------------------------------------------------------------------------
// Grid-wide barrier (deterministic generation target, one bump per barrier).
// ---------------------------------------------------------------------------
__device__ __forceinline__ void grid_sync(unsigned* bar, unsigned* gen, unsigned target) {
  __syncthreads();
  if (threadIdx.x == 0) {
    __threadfence();
    unsigned old = __atomic_fetch_add(bar, 1u, __ATOMIC_ACQ_REL);
    if (old == NWG - 1) {
      __atomic_store_n(bar, 0u, __ATOMIC_RELAXED);
      __atomic_fetch_add(gen, 1u, __ATOMIC_ACQ_REL);
    } else {
      while (__atomic_load_n(gen, __ATOMIC_ACQUIRE) < target)
        __builtin_amdgcn_s_sleep(2);
    }
    __threadfence();
  }
  __syncthreads();
}

// ---------------------------------------------------------------------------
// Fused persistent recurrence, software pipelined (one grid barrier / step).
// LDS (320KB = full WGP):
//   192KB weight fragment slices (persistent), 2x64KB h staging (TDM per step).
// Phase A (h0@Whh0^T) and Phase B GEMM1 (h0@Wih1^T) share A-fragments ->
// fused k-loop with 4 independent WMMA accumulators.
// ---------------------------------------------------------------------------
#define SW_HALFS   (3 * 4 * 16 * 512)          // 98304 halfs = 192KB
#define SH_HALFS   (BATCH * HID)               // 32768 halfs = 64KB
#define SMEM_HALFS (SW_HALFS + 2 * SH_HALFS)   // 163840 halfs = 320KB
#define SH0_BYTE_OFF (SW_HALFS * 2)            // 196608
#define SH1_BYTE_OFF (SW_HALFS * 2 + SH_HALFS * 2)  // 262144

__device__ __forceinline__ v16h lds_b_frag(const _Float16* sW, int mat, int tjl,
                                           int kt, int lane) {
  return *(const v16h*)(sW + (((mat * 4 + tjl) * 16 + kt) * 512) + lane * 16);
}

__global__ __launch_bounds__(NTHREADS) void prune_rnn_recur(
    const _Float16* __restrict__ xb0, const _Float16* __restrict__ fWhh0,
    const _Float16* __restrict__ fWih1, const _Float16* __restrict__ fWhh1,
    const float* __restrict__ bias1, _Float16* __restrict__ h0,
    _Float16* __restrict__ h1, float* __restrict__ out,
    unsigned* bar, unsigned* gen) {
  __shared__ _Float16 smem[SMEM_HALFS];
  _Float16* sW  = smem;
  _Float16* sH0 = smem + SW_HALFS;
  _Float16* sH1 = smem + SW_HALFS + SH_HALFS;

  const int lane = threadIdx.x & 31;
  const int w = threadIdx.x >> 5;
  const int m = w & 3;
  const int tjl0 = (w >> 2) * 2;
  const int tjl1 = tjl0 + 1;
  const int KT = HID / 32;  // 16
  const int Mrow = m * 16 + (lane & 15);
  const int kboff = (lane & 16) ? 8 : 0;
  const int rb = (lane & 16) ? 8 : 0;
  const int jc0 = (blockIdx.x * 4 + tjl0) * 16 + (lane & 15);
  const int jc1 = jc0 + 16;
  const size_t HB = (size_t)BATCH * HID;

  // one-time: stage this WG's weight fragment slices into LDS (contiguous)
  {
    const size_t slice = (size_t)blockIdx.x * 4 * 16 * 512;
    const _Float16* srcs[3] = {fWhh0 + slice, fWih1 + slice, fWhh1 + slice};
#pragma unroll
    for (int mat = 0; mat < 3; ++mat) {
      const v8h* src = (const v8h*)srcs[mat];
      v8h* dst = (v8h*)(sW + mat * (4 * 16 * 512));
      for (int i = threadIdx.x; i < (4 * 16 * 512) / 8; i += NTHREADS) dst[i] = src[i];
    }
  }

  const float bj0 = bias1[jc0];  // t-invariant
  const float bj1 = bias1[jc1];

  for (int t = 0; t <= S_LEN; ++t) {
    // ---- TDM: DMA h0[t-1] and h1[t-2] into LDS (wave 0 issues) ----
    if (threadIdx.x < 32) {
      tdm_load_1d(SH0_BYTE_OFF, h0 + (size_t)((t ^ 1) & 1) * HB, SH_HALFS * 2);
      tdm_load_1d(SH1_BYTE_OFF, h1 + (size_t)(t & 1) * HB, SH_HALFS * 2);
      __builtin_amdgcn_s_wait_tensorcnt(0);
    }
    __syncthreads();

    const _Float16* arowH0 = sH0 + (size_t)Mrow * HID;
    const _Float16* arowH1 = sH1 + (size_t)Mrow * HID;

    // hoist xb0 additive terms so their L2 latency overlaps the GEMMs
    float xv0[8], xv1[8];
    if (t < S_LEN) {
      const _Float16* xrow = xb0 + (size_t)t * HB;
#pragma unroll
      for (int r = 0; r < 8; ++r) {
        int bi = m * 16 + rb + r;
        xv0[r] = (float)xrow[(size_t)bi * HID + jc0];
        xv1[r] = (float)xrow[(size_t)bi * HID + jc1];
      }
      if (t + 1 < S_LEN)
        __builtin_prefetch(xb0 + (size_t)(t + 1) * HB + (size_t)Mrow * HID, 0, 0);
    }

    v8f accA0 = {}, accA1 = {}, accB0 = {}, accB1 = {};

    if (t > 0 && t < S_LEN) {
      // fused: one A-fragment feeds Whh0 (phase A) and Wih1 (phase B) tiles
#pragma unroll 4
      for (int kt = 0; kt < KT; ++kt) {
        v16h a   = load_a_f16(arowH0, kt * 32 + kboff);
        v16h b00 = lds_b_frag(sW, 0, tjl0, kt, lane);
        v16h b01 = lds_b_frag(sW, 0, tjl1, kt, lane);
        v16h b10 = lds_b_frag(sW, 1, tjl0, kt, lane);
        v16h b11 = lds_b_frag(sW, 1, tjl1, kt, lane);
        accA0 = wmma16(a, b00, accA0);
        accA1 = wmma16(a, b01, accA1);
        accB0 = wmma16(a, b10, accB0);
        accB1 = wmma16(a, b11, accB1);
      }
#pragma unroll 4
      for (int kt = 0; kt < KT; ++kt) {
        v16h a  = load_a_f16(arowH1, kt * 32 + kboff);
        v16h b0 = lds_b_frag(sW, 2, tjl0, kt, lane);
        v16h b1 = lds_b_frag(sW, 2, tjl1, kt, lane);
        accB0 = wmma16(a, b0, accB0);
        accB1 = wmma16(a, b1, accB1);
      }
    } else if (t == 0) {
#pragma unroll 4
      for (int kt = 0; kt < KT; ++kt) {
        v16h a  = load_a_f16(arowH0, kt * 32 + kboff);
        v16h b0 = lds_b_frag(sW, 0, tjl0, kt, lane);
        v16h b1 = lds_b_frag(sW, 0, tjl1, kt, lane);
        accA0 = wmma16(a, b0, accA0);
        accA1 = wmma16(a, b1, accA1);
      }
    } else {  // t == S_LEN: only phase B remains
#pragma unroll 4
      for (int kt = 0; kt < KT; ++kt) {
        v16h a  = load_a_f16(arowH0, kt * 32 + kboff);
        v16h b0 = lds_b_frag(sW, 1, tjl0, kt, lane);
        v16h b1 = lds_b_frag(sW, 1, tjl1, kt, lane);
        accB0 = wmma16(a, b0, accB0);
        accB1 = wmma16(a, b1, accB1);
      }
#pragma unroll 4
      for (int kt = 0; kt < KT; ++kt) {
        v16h a  = load_a_f16(arowH1, kt * 32 + kboff);
        v16h b0 = lds_b_frag(sW, 2, tjl0, kt, lane);
        v16h b1 = lds_b_frag(sW, 2, tjl1, kt, lane);
        accB0 = wmma16(a, b0, accB0);
        accB1 = wmma16(a, b1, accB1);
      }
    }

    // ---- epilogue A: h0[t] = tanh(xb0[t] + accA) ----
    if (t < S_LEN) {
      _Float16* h0cur = h0 + (size_t)(t & 1) * HB;
#pragma unroll
      for (int r = 0; r < 8; ++r) {
        int bi = m * 16 + rb + r;
        float v0 = tanhf(accA0[r] + xv0[r]);
        float v1 = tanhf(accA1[r] + xv1[r]);
        h0cur[(size_t)bi * HID + jc0] = (_Float16)v0;
        h0cur[(size_t)bi * HID + jc1] = (_Float16)v1;
      }
    }
    // ---- epilogue B: h1[t-1] = tanh(accB + b1) ----
    if (t > 0) {
      const int u = t - 1;
      _Float16* h1cur = h1 + (size_t)(u & 1) * HB;
#pragma unroll
      for (int r = 0; r < 8; ++r) {
        int bi = m * 16 + rb + r;
        float v0 = tanhf(accB0[r] + bj0);
        float v1 = tanhf(accB1[r] + bj1);
        h1cur[(size_t)bi * HID + jc0] = (_Float16)v0;
        h1cur[(size_t)bi * HID + jc1] = (_Float16)v1;
        if (u == S_LEN - 1) {
          out[(size_t)bi * HID + jc0] = v0;
          out[(size_t)bi * HID + jc1] = v1;
        }
      }
    }
    grid_sync(bar, gen, (unsigned)(t + 1));
  }
}

// ---------------------------------------------------------------------------
extern "C" void kernel_launch(void* const* d_in, const int* in_sizes, int n_in,
                              void* d_out, int out_size, void* d_ws, size_t ws_size,
                              hipStream_t stream) {
  const float* x    = (const float*)d_in[0];
  const float* Wih0 = (const float*)d_in[1];
  const float* Whh0 = (const float*)d_in[2];
  const float* bih0 = (const float*)d_in[3];
  const float* bhh0 = (const float*)d_in[4];
  const float* Wih1 = (const float*)d_in[5];
  const float* Whh1 = (const float*)d_in[6];
  const float* bih1 = (const float*)d_in[7];
  const float* bhh1 = (const float*)d_in[8];
  const int* mih0 = (const int*)d_in[9];
  const int* mhh0 = (const int*)d_in[10];
  const int* mih1 = (const int*)d_in[11];
  const int* mhh1 = (const int*)d_in[12];
  float* out = (float*)d_out;

  char* base = (char*)d_ws;
  size_t off = 0;
  auto carve = [&](size_t bytes) -> void* {
    void* p = (void*)(base + off);
    off += (bytes + 63) & ~(size_t)63;
    return p;
  };
  unsigned* bar   = (unsigned*)carve(8);
  unsigned* gen   = (unsigned*)((char*)bar + 4);
  float* bias0    = (float*)carve(HID * sizeof(float));
  float* bias1    = (float*)carve(HID * sizeof(float));
  _Float16* fWih0 = (_Float16*)carve((size_t)HID * IN_DIM * 2);
  _Float16* fWhh0 = (_Float16*)carve((size_t)HID * HID * 2);
  _Float16* fWih1 = (_Float16*)carve((size_t)HID * HID * 2);
  _Float16* fWhh1 = (_Float16*)carve((size_t)HID * HID * 2);
  _Float16* h0    = (_Float16*)carve((size_t)2 * BATCH * HID * 2);
  _Float16* h1    = (_Float16*)carve((size_t)2 * BATCH * HID * 2);
  _Float16* xb0   = (_Float16*)carve((size_t)S_LEN * BATCH * HID * 2);
  (void)ws_size; (void)in_sizes; (void)n_in; (void)out_size;

  prune_prep_weights<<<(HID * IN_DIM + 255) / 256, 256, 0, stream>>>(Wih0, mih0, fWih0, IN_DIM);
  prune_prep_weights<<<(HID * HID + 255) / 256, 256, 0, stream>>>(Whh0, mhh0, fWhh0, HID);
  prune_prep_weights<<<(HID * HID + 255) / 256, 256, 0, stream>>>(Wih1, mih1, fWih1, HID);
  prune_prep_weights<<<(HID * HID + 255) / 256, 256, 0, stream>>>(Whh1, mhh1, fWhh1, HID);

  prune_prep_misc<<<(2 * BATCH * HID + 255) / 256, 256, 0, stream>>>(
      bih0, bhh0, bih1, bhh1, bias0, bias1, h0, h1, bar, gen);

  prune_xb0_gemm<<<S_LEN, NTHREADS, 0, stream>>>(x, fWih0, bias0, xb0);

  prune_rnn_recur<<<NWG, NTHREADS, 0, stream>>>(xb0, fWhh0, fWih1, fWhh1,
                                                bias1, h0, h1, out, bar, gen);
}
